// StructurePredictor_32229434589571
// MI455X (gfx1250) — matrix operand order repaired and checked by reading, not compiled, not run on hardware
//
#include <hip/hip_runtime.h>
#include <hip/hip_bf16.h>
#include <cmath>

// ---------------- problem dims ----------------
static constexpr int Bz = 4;
static constexpr int Sz = 1024;
static constexpr int Ez = 768;
static constexpr int Hh = 8;
static constexpr int HDd = 96;
static constexpr int Mrows = Bz * Sz;       // 4096

typedef __attribute__((ext_vector_type(16))) __bf16 bf16x16;
typedef __attribute__((ext_vector_type(8)))  float  floatx8;

union frag_u  { bf16x16 v; uint4 q[2]; };          // 32 B
union pack16  { __bf16 h[16]; uint4 q[2]; };       // 32 B
union pack32  { __bf16 h[32]; uint4 q[4]; };       // 64 B

// =====================================================================
// Tiled WMMA GEMM:  C[M,N] = act(A[M,K] @ W[K,N] + bias[N])
// N, K are COMPILE-TIME so all strided addresses fold into the 24-bit
// instruction offsets (one address VGPR + s_clause of immediate-offset
// loads; no per-load 64-bit adds / xcnt stalls).
// 128 threads = 4 waves; block tile 64x128; each wave 32x64 (2x4 WMMA).
// A staged [m][k] bf16; W staged TRANSPOSED [n][k] bf16 so every WMMA
// fragment is contiguous b128 LDS traffic. f32 accumulate.
// =====================================================================
template<int N, int K, int RELU>
__global__ __launch_bounds__(128)
void wmma_gemm_bias(const float* __restrict__ A, const float* __restrict__ W,
                    const float* __restrict__ bias, float* __restrict__ C)
{
    __shared__ __attribute__((aligned(64))) __bf16 As[64][32];    // [m][k] 4 KB
    __shared__ __attribute__((aligned(64))) __bf16 Bs[128][32];   // [n][k] 8 KB (transposed)

    const int tid  = threadIdx.x;
    const int wave = tid >> 5;
    const int lane = tid & 31;
    const int wr = wave >> 1;        // 0..1 -> 32 rows
    const int wc = wave & 1;         // 0..1 -> 64 cols
    const int m0 = blockIdx.y * 64;
    const int n0 = blockIdx.x * 128;

    const floatx8 zero8 = {0.f,0.f,0.f,0.f,0.f,0.f,0.f,0.f};
    floatx8 acc[2][4];
    #pragma unroll
    for (int i = 0; i < 2; ++i)
        #pragma unroll
        for (int j = 0; j < 4; ++j) acc[i][j] = zero8;

    const int ar  = tid >> 1;               // A stage: row, 2 threads/row
    const int ac0 = (tid & 1) * 16;         //          16-elem half
    const int bn  = tid;                    // B stage: one n-column per thread

    const float* asrc = A + (size_t)(m0 + ar) * K + ac0;   // advance by 32 per chunk
    const float* bsrc = W + n0 + bn;                       // advance by 32*N per chunk

    const int mrow = lane & 15;
    const int g    = lane >> 4;

    #pragma unroll 1
    for (int kk = 0; kk < K; kk += 32) {
        // ---- stage A tile 64x32: coalesced b128 loads, packed b128 stores ----
        {
            pack16 ta;
            #pragma unroll
            for (int c = 0; c < 16; ++c) ta.h[c] = (__bf16)asrc[c];
            uint4* dst = (uint4*)&As[ar][ac0];
            dst[0] = ta.q[0]; dst[1] = ta.q[1];
        }
        // ---- stage W tile 32x128 TRANSPOSED: thread owns column n=bn ----
        // lane-consecutive n => coalesced b32 with immediate offsets k*N*4
        {
            pack32 tb;
            #pragma unroll
            for (int k = 0; k < 32; ++k) tb.h[k] = (__bf16)bsrc[k * N];
            uint4* dst = (uint4*)&Bs[bn][0];
            dst[0] = tb.q[0]; dst[1] = tb.q[1]; dst[2] = tb.q[2]; dst[3] = tb.q[3];
        }
        // prefetch next K-chunk (global_prefetch_b8)
        if (kk + 32 < K) {
            __builtin_prefetch(asrc + 32, 0, 1);
            __builtin_prefetch(bsrc + 32 * N, 0, 1);
        }
        asrc += 32;
        bsrc += (size_t)32 * N;
        __syncthreads();

        // ---- A fragments: two contiguous 16B runs (k = g*8.., k = 16+g*8..) ----
        frag_u afrag[2];
        #pragma unroll
        for (int i = 0; i < 2; ++i) {
            const uint4* arow = (const uint4*)&As[wr * 32 + i * 16 + mrow][0];
            afrag[i].q[0] = arow[g];       // k = g*8 .. g*8+7
            afrag[i].q[1] = arow[2 + g];   // k = 16+g*8 .. 16+g*8+7
        }
        // ---- B fragments: two contiguous 16B runs (k = g*8.., k = 16+g*8..) ----
        frag_u bfrag[4];
        #pragma unroll
        for (int j = 0; j < 4; ++j) {
            const uint4* brow = (const uint4*)&Bs[wc * 64 + j * 16 + mrow][0];
            bfrag[j].q[0] = brow[g];
            bfrag[j].q[1] = brow[2 + g];
        }

        #pragma unroll
        for (int i = 0; i < 2; ++i)
            #pragma unroll
            for (int j = 0; j < 4; ++j)
                acc[i][j] = __builtin_amdgcn_wmma_f32_16x16x32_bf16(
                    false, afrag[i].v, false, bfrag[j].v,
                    (short)0, acc[i][j], false, false);
        __syncthreads();
    }

    // ---- epilogue: bias (+relu), store.  C/D layout: m = r + 8*(lane>=16) ----
    const int lcol = lane & 15;
    const int lhi  = lane >> 4;
    #pragma unroll
    for (int i = 0; i < 2; ++i) {
        #pragma unroll
        for (int j = 0; j < 4; ++j) {
            const int col = n0 + wc * 64 + j * 16 + lcol;
            const float bv = bias[col];
            #pragma unroll
            for (int r = 0; r < 8; ++r) {
                const int row = m0 + wr * 32 + i * 16 + lhi * 8 + r;
                float x = acc[i][j][r] + bv;
                if (RELU) x = fmaxf(x, 0.f);
                C[(size_t)row * N + col] = x;
            }
        }
    }
}

// =====================================================================
// Attention over the BATCH axis (torch batch_first=False quirk).
// One thread per (s,h): 4x4 score matrix over HD=96, softmax over c.
// =====================================================================
__global__ void attn_over_batch(const float* __restrict__ q,
                                const float* __restrict__ k,
                                const float* __restrict__ v,
                                float* __restrict__ ao)
{
    int idx = blockIdx.x * blockDim.x + threadIdx.x;
    if (idx >= Sz * Hh) return;
    const int s = idx / Hh, h = idx % Hh;
    const int off = h * HDd;
    const float scale = rsqrtf((float)HDd);

    float sc[Bz][Bz];
    for (int b = 0; b < Bz; ++b) {
        const float* qp = q + ((size_t)b * Sz + s) * Ez + off;
        for (int c = 0; c < Bz; ++c) {
            const float* kp = k + ((size_t)c * Sz + s) * Ez + off;
            float d = 0.f;
            for (int t = 0; t < HDd; ++t) d += qp[t] * kp[t];
            sc[b][c] = d * scale;
        }
    }
    for (int b = 0; b < Bz; ++b) {
        float mx = sc[b][0];
        for (int c = 1; c < Bz; ++c) mx = fmaxf(mx, sc[b][c]);
        float sum = 0.f;
        for (int c = 0; c < Bz; ++c) { sc[b][c] = __expf(sc[b][c] - mx); sum += sc[b][c]; }
        float inv = 1.f / sum;
        for (int c = 0; c < Bz; ++c) sc[b][c] *= inv;
    }
    for (int b = 0; b < Bz; ++b) {
        float* op = ao + ((size_t)b * Sz + s) * Ez + off;
        for (int t = 0; t < HDd; ++t) {
            float acc = 0.f;
            for (int c = 0; c < Bz; ++c)
                acc += sc[b][c] * v[((size_t)c * Sz + s) * Ez + off + t];
            op[t] = acc;
        }
    }
}

// probs[r] = m1[r,:] . w[:,0] + b        (r = b*S+s, 4096 rows, K=384)
__global__ void probs_kernel(const float* __restrict__ m1,
                             const float* __restrict__ w,
                             const float* __restrict__ b2,
                             float* __restrict__ probs)
{
    int r = blockIdx.x * blockDim.x + threadIdx.x;
    if (r >= Mrows) return;
    const float* row = m1 + (size_t)r * 384;
    float acc = b2[0];
    for (int t = 0; t < 384; ++t) acc += row[t] * w[t];
    probs[r] = acc;
}

// contacts[b,s,j] = probs[b*S+s]; S=1024 so index>>10 recovers (b*S+s)
__global__ void fill_contacts(const float* __restrict__ probs, float* __restrict__ contacts)
{
    size_t i = (size_t)blockIdx.x * blockDim.x + threadIdx.x;
    if (i >= (size_t)Bz * Sz * Sz) return;
    contacts[i] = probs[i >> 10];
}

// steps from rb head: ba = hb @ rb_w2 + rb_b2 ; steps = 3.8*{cpcq, spcq, sq}; row s==0 -> 0
__global__ void steps_kernel(const float* __restrict__ hb,
                             const float* __restrict__ w,   // [384,3]
                             const float* __restrict__ bias,
                             float* __restrict__ steps)
{
    int r = blockIdx.x * blockDim.x + threadIdx.x;
    if (r >= Mrows) return;
    float a0 = bias[0], a1 = bias[1];
    const float* row = hb + (size_t)r * 384;
    for (int t = 0; t < 384; ++t) {
        float h = row[t];
        a0 += h * w[t * 3 + 0];
        a1 += h * w[t * 3 + 1];
    }
    const int s = r & (Sz - 1);
    float st0 = 0.f, st1 = 0.f, st2 = 0.f;
    if (s != 0) {
        float cp = cosf(a0), sp = sinf(a0), cq = cosf(a1), sq = sinf(a1);
        st0 = 3.8f * cp * cq; st1 = 3.8f * sp * cq; st2 = 3.8f * sq;
    }
    steps[(size_t)r * 3 + 0] = st0;
    steps[(size_t)r * 3 + 1] = st1;
    steps[(size_t)r * 3 + 2] = st2;
}

// cumsum over S per (b,dim) — 12 serial chains, trivially small
__global__ void cumsum_kernel(const float* __restrict__ steps, float* __restrict__ structure)
{
    int t = threadIdx.x;
    if (t >= Bz * 3) return;
    const int b = t / 3, d = t % 3;
    float acc = 0.f;
    for (int s = 0; s < Sz; ++s) {
        acc += steps[((size_t)b * Sz + s) * 3 + d];
        structure[((size_t)b * Sz + s) * 3 + d] = acc;
    }
}

__global__ void zero_kernel(float* __restrict__ p, int n)
{
    int i = blockIdx.x * blockDim.x + threadIdx.x;
    if (i < n) p[i] = 0.f;
}

// Gradient of contact loss wrt structure.
// g_i = (1/(B*S*S)) * sum_j sign(d_ij-8) * (p_i+p_j) * (s_i-s_j)/d_ij
// (chain loss has zero gradient: diagonal distances are identically 0)
__global__ __launch_bounds__(256)
void grad_kernel(const float* __restrict__ s, const float* __restrict__ probs,
                 float* __restrict__ g)
{
    __shared__ float sx[Sz], sy[Sz], sz[Sz], pr[Sz];
    const int b = blockIdx.x;
    const int tid = threadIdx.x;
    for (int i = tid; i < Sz; i += 256) {
        sx[i] = s[((size_t)b * Sz + i) * 3 + 0];
        sy[i] = s[((size_t)b * Sz + i) * 3 + 1];
        sz[i] = s[((size_t)b * Sz + i) * 3 + 2];
        pr[i] = probs[(size_t)b * Sz + i];
    }
    __syncthreads();
    const float inv = 1.0f / ((float)Bz * (float)Sz * (float)Sz);
    for (int i = tid; i < Sz; i += 256) {
        const float xi = sx[i], yi = sy[i], zi = sz[i], pi = pr[i];
        float gx = 0.f, gy = 0.f, gz = 0.f;
        for (int j = 0; j < Sz; ++j) {
            float dx = xi - sx[j], dy = yi - sy[j], dz = zi - sz[j];
            float d2 = dx * dx + dy * dy + dz * dz;
            if (d2 > 0.f) {
                float d = sqrtf(d2);
                float sgn = (d > 8.f) ? 1.f : ((d < 8.f) ? -1.f : 0.f);
                float w = sgn * (pi + pr[j]) / d;
                gx += w * dx; gy += w * dy; gz += w * dz;
            }
        }
        g[((size_t)b * Sz + i) * 3 + 0] = gx * inv;
        g[((size_t)b * Sz + i) * 3 + 1] = gy * inv;
        g[((size_t)b * Sz + i) * 3 + 2] = gz * inv;
    }
}

__global__ void adam_kernel(float* __restrict__ s, float* __restrict__ m,
                            float* __restrict__ v, const float* __restrict__ g,
                            float c1, float c2)   // c1 = 1/(1-b1^t), c2 = 1/(1-b2^t)
{
    int i = blockIdx.x * blockDim.x + threadIdx.x;
    if (i >= Bz * Sz * 3) return;
    float gi = g[i];
    float mi = 0.9f   * m[i] + 0.1f   * gi;
    float vi = 0.999f * v[i] + 0.001f * gi * gi;
    m[i] = mi; v[i] = vi;
    float mh = mi * c1, vh = vi * c2;
    s[i] -= 0.01f * mh / (sqrtf(vh) + 1e-8f);
}

// =====================================================================
extern "C" void kernel_launch(void* const* d_in, const int* in_sizes, int n_in,
                              void* d_out, int out_size, void* d_ws, size_t ws_size,
                              hipStream_t stream)
{
    const float* x      = (const float*)d_in[0];
    const float* bb_w1  = (const float*)d_in[1];  const float* bb_b1 = (const float*)d_in[2];
    const float* bb_w2  = (const float*)d_in[3];  const float* bb_b2 = (const float*)d_in[4];
    const float* sc_w1  = (const float*)d_in[5];  const float* sc_b1 = (const float*)d_in[6];
    const float* sc_w2  = (const float*)d_in[7];  const float* sc_b2 = (const float*)d_in[8];
    const float* wq = (const float*)d_in[9];  const float* bq = (const float*)d_in[10];
    const float* wk = (const float*)d_in[11]; const float* bk = (const float*)d_in[12];
    const float* wv = (const float*)d_in[13]; const float* bv = (const float*)d_in[14];
    const float* wo = (const float*)d_in[15]; const float* bo = (const float*)d_in[16];
    const float* mlp_w1 = (const float*)d_in[17]; const float* mlp_b1 = (const float*)d_in[18];
    const float* mlp_w2 = (const float*)d_in[19]; const float* mlp_b2 = (const float*)d_in[20];
    const float* rb_w1  = (const float*)d_in[21]; const float* rb_b1  = (const float*)d_in[22];
    const float* rb_w2  = (const float*)d_in[23]; const float* rb_b2  = (const float*)d_in[24];
    // rs_* (25..28) are computed-then-discarded in the reference: skipped.

    float* out = (float*)d_out;
    const size_t mat = (size_t)Mrows * Ez;                   // 3,145,728
    float* backbone  = out;                                  // [B,S,E]
    float* side      = out + mat;                            // [B,S,E]
    float* contacts  = out + 2 * mat;                        // [B,S,S]
    float* structure = out + 2 * mat + (size_t)Bz * Sz * Sz; // [B,S,3]

    // ---- workspace layout (floats) ----
    float* ws    = (float*)d_ws;
    float* qb    = ws;                 // 3,145,728
    float* kb    = qb + mat;
    float* vb    = kb + mat;
    float* tA    = vb + mat;           // t1 / ao
    float* tB    = tA + mat;           // aow
    float* h384  = tB + mat;           // 4096x384 (t2 / m1 / rb hidden)
    float* probs = h384 + (size_t)Mrows * 384;
    float* steps = probs + Mrows;      // 4096x3
    float* grad  = steps + (size_t)Mrows * 3;
    float* am    = grad + (size_t)Bz * Sz * 3;
    float* av    = am + (size_t)Bz * Sz * 3;

    const dim3 blk128(128);
    const dim3 gN768(Ez / 128, Mrows / 64);    // (6, 64)
    const dim3 gN384(384 / 128, Mrows / 64);   // (3, 64)

    // backbone / side feature networks
    wmma_gemm_bias<768, 768, 1><<<gN768, blk128, 0, stream>>>(x,  bb_w1, bb_b1, tA);
    wmma_gemm_bias<768, 768, 0><<<gN768, blk128, 0, stream>>>(tA, bb_w2, bb_b2, backbone);
    wmma_gemm_bias<384, 768, 1><<<gN384, blk128, 0, stream>>>(backbone, sc_w1, sc_b1, h384);
    wmma_gemm_bias<768, 384, 0><<<gN768, blk128, 0, stream>>>(h384, sc_w2, sc_b2, side);

    // contact predictor: QKV projections, batch-axis attention, out proj, MLP
    wmma_gemm_bias<768, 768, 0><<<gN768, blk128, 0, stream>>>(x, wq, bq, qb);
    wmma_gemm_bias<768, 768, 0><<<gN768, blk128, 0, stream>>>(x, wk, bk, kb);
    wmma_gemm_bias<768, 768, 0><<<gN768, blk128, 0, stream>>>(x, wv, bv, vb);
    attn_over_batch<<<(Sz * Hh + 127) / 128, 128, 0, stream>>>(qb, kb, vb, tA);
    wmma_gemm_bias<768, 768, 0><<<gN768, blk128, 0, stream>>>(tA, wo, bo, tB);
    wmma_gemm_bias<384, 768, 1><<<gN384, blk128, 0, stream>>>(tB, mlp_w1, mlp_b1, h384);
    probs_kernel<<<(Mrows + 255) / 256, 256, 0, stream>>>(h384, mlp_w2, mlp_b2, probs);
    {
        size_t tot = (size_t)Bz * Sz * Sz;
        fill_contacts<<<(unsigned)((tot + 255) / 256), 256, 0, stream>>>(probs, contacts);
    }

    // structure refiner init
    wmma_gemm_bias<384, 768, 1><<<gN384, blk128, 0, stream>>>(backbone, rb_w1, rb_b1, h384);
    steps_kernel<<<(Mrows + 255) / 256, 256, 0, stream>>>(h384, rb_w2, rb_b2, steps);
    cumsum_kernel<<<1, 32, 0, stream>>>(steps, structure);

    // 50 Adam steps on structure
    zero_kernel<<<(2 * Bz * Sz * 3 + 255) / 256, 256, 0, stream>>>(am, 2 * Bz * Sz * 3);
    for (int t = 1; t <= 50; ++t) {
        float c1 = 1.0f / (1.0f - powf(0.9f,   (float)t));
        float c2 = 1.0f / (1.0f - powf(0.999f, (float)t));
        grad_kernel<<<Bz, 256, 0, stream>>>(structure, probs, grad);
        adam_kernel<<<(Bz * Sz * 3 + 255) / 256, 256, 0, stream>>>(structure, am, av, grad, c1, c2);
    }
}
